// Attention_56169582297517
// MI455X (gfx1250) — matrix operand order repaired and checked by reading, not compiled, not run on hardware
//
#include <hip/hip_runtime.h>
#include <hip/hip_bf16.h>

typedef __attribute__((ext_vector_type(16))) _Float16 v16h;
typedef __attribute__((ext_vector_type(8)))  float    v8f;

union H16 { uint4 u[2]; v16h h; };

#define WMMA_F16(A, B, C) \
    __builtin_amdgcn_wmma_f32_16x16x32_f16(false, (A), false, (B), (short)0, (C), false, false)

static constexpr int   E_DIM = 512;
static constexpr int   L_DIM = 128;
static constexpr float EPSV  = 1e-12f;

// ---------------- W f32 -> f16 ----------------
__global__ void cvt_w_kernel(const float* __restrict__ W, _Float16* __restrict__ Wh, int n)
{
    int i = blockIdx.x * 256 + threadIdx.x;
    if (i < n) Wh[i] = (_Float16)W[i];
}

// ------- secondary [M,E] f32 -> ST [E,M] f16 (transposed, for B fragments) -------
__global__ __launch_bounds__(256)
void transpose_cvt_kernel(const float* __restrict__ S, _Float16* __restrict__ T, int M)
{
    __shared__ float tile[64][65];
    int m0 = blockIdx.x * 64, e0 = blockIdx.y * 64;
    int t = threadIdx.x;
#pragma unroll
    for (int i = 0; i < 16; ++i) {
        int idx = t + i * 256;
        int r = idx >> 6, c = idx & 63;
        tile[r][c] = S[(size_t)(m0 + r) * E_DIM + e0 + c];
    }
    __syncthreads();
    int e = t >> 2, mc = (t & 3) * 16;
    H16 o;
#pragma unroll
    for (int j = 0; j < 16; ++j) o.h[j] = (_Float16)tile[mc + j][e];
    _Float16* dst = T + (size_t)(e0 + e) * M + m0 + mc;
    *(uint4*)(dst)     = o.u[0];
    *(uint4*)(dst + 8) = o.u[1];
}

// A-fragment (16x32 f16, M = lane%16, K = (j/8)*16 + (lane/16)*8 + j%8) built from f32 row.
// ap must already point at row_start + kb + (lane/16)*8.
__device__ inline v16h load_a_from_f32(const float* __restrict__ ap)
{
    float4 f0 = *(const float4*)(ap);
    float4 f1 = *(const float4*)(ap + 4);
    float4 f2 = *(const float4*)(ap + 16);
    float4 f3 = *(const float4*)(ap + 20);
    v16h h;
    h[0]=(_Float16)f0.x;  h[1]=(_Float16)f0.y;  h[2]=(_Float16)f0.z;  h[3]=(_Float16)f0.w;
    h[4]=(_Float16)f1.x;  h[5]=(_Float16)f1.y;  h[6]=(_Float16)f1.z;  h[7]=(_Float16)f1.w;
    h[8]=(_Float16)f2.x;  h[9]=(_Float16)f2.y;  h[10]=(_Float16)f2.z; h[11]=(_Float16)f2.w;
    h[12]=(_Float16)f3.x; h[13]=(_Float16)f3.y; h[14]=(_Float16)f3.z; h[15]=(_Float16)f3.w;
    return h;
}

// ---------------- q = primary@W^T + b ; k = secondary@W^T + b  (f16 out) ----------------
__global__ __launch_bounds__(256)
void proj_kernel(const float* __restrict__ X0, const float* __restrict__ X1,
                 const _Float16* __restrict__ Wh, const float* __restrict__ bias,
                 _Float16* __restrict__ Q, _Float16* __restrict__ K, int N, int M)
{
    int blk = blockIdx.x;
    const float* X; _Float16* D; int n0;
    if (blk * 32 < N) { X = X0; D = Q; n0 = blk * 32; }
    else              { X = X1; D = K; n0 = blk * 32 - N; }

    int lane = threadIdx.x & 31;
    int w    = threadIdx.x >> 5;   // 8 waves, each owns 16 L-columns
    int half = lane >> 4, l16 = lane & 15;

    v8f vz = {};
    v8f acc0 = vz, acc1 = vz;      // tile rows 0,1 (16 rows each)

    const float* r0 = X + (size_t)(n0 + 0  + l16) * E_DIM;
    const float* r1 = X + (size_t)(n0 + 16 + l16) * E_DIM;
    const _Float16* wrow = Wh + (size_t)(w * 16 + l16) * E_DIM;

#pragma unroll 4
    for (int kb = 0; kb < E_DIM; kb += 32) {
        H16 bf;
        const _Float16* bp = wrow + kb + half * 8;
        bf.u[0] = *(const uint4*)(bp);
        bf.u[1] = *(const uint4*)(bp + 16);
        v16h a0 = load_a_from_f32(r0 + kb + half * 8);
        v16h a1 = load_a_from_f32(r1 + kb + half * 8);
        acc0 = WMMA_F16(a0, bf.h, acc0);
        acc1 = WMMA_F16(a1, bf.h, acc1);
    }

    float bl = bias[w * 16 + l16];
#pragma unroll
    for (int r = 0; r < 8; ++r) {
        int row0 = n0 + r + 8 * half;
        D[(size_t)(row0)      * L_DIM + w * 16 + l16] = (_Float16)(acc0[r] + bl);
        D[(size_t)(row0 + 16) * L_DIM + w * 16 + l16] = (_Float16)(acc1[r] + bl);
    }
}

// ---------------- fused: relu(q k^T) -> sumsq + context, normalize ----------------
// Block: 32 output rows x 256 E-cols, 8 waves. M-tile = 128, double-buffered S in LDS.
__global__ __launch_bounds__(256)
void fused_attn_kernel(const _Float16* __restrict__ Q, const _Float16* __restrict__ Kh,
                       const _Float16* __restrict__ ST, float* __restrict__ Out,
                       int N, int M)
{
    constexpr int SSTR = 136;                 // 128 + 8 pad (keeps 16B alignment)
    __shared__ _Float16 Sl[2][32 * SSTR];
    __shared__ float    sumsq[32];

    int n0 = blockIdx.x * 32;
    int eb = blockIdx.y * 256;
    int tid = threadIdx.x, lane = tid & 31, w = tid >> 5;
    int half = lane >> 4, l16 = lane & 15;

    if (tid < 32) sumsq[tid] = 0.f;
    __syncthreads();

    int e0w = eb + w * 32;                    // phase-2: 32 E-columns per wave

    // Preload q A-fragments (invariant across the whole M loop): 2 tile-rows x 4 K-chunks.
    v16h qa[2][4];
#pragma unroll
    for (int ti = 0; ti < 2; ++ti) {
        const _Float16* qb = Q + (size_t)(n0 + ti * 16 + l16) * L_DIM + half * 8;
#pragma unroll
        for (int c = 0; c < 4; ++c) {
            H16 t;
            t.u[0] = *(const uint4*)(qb + c * 32);
            t.u[1] = *(const uint4*)(qb + c * 32 + 16);
            qa[ti][c] = t.h;
        }
    }

    v8f vz = {};
    v8f acc[4];                               // [ti*2 + tc]
#pragma unroll
    for (int i = 0; i < 4; ++i) acc[i] = vz;
    float sq[2][8];
#pragma unroll
    for (int ti = 0; ti < 2; ++ti)
#pragma unroll
        for (int r = 0; r < 8; ++r) sq[ti][r] = 0.f;

    int iters = M / 128;
    for (int it = 0; it < iters; ++it) {
        int m0 = it * 128;
        _Float16* Sbuf = Sl[it & 1];

        // ---- phase 1: S = relu(q @ k^T), [32 x 128]; wave w owns cols [w*16, w*16+16)
        v8f d0 = vz, d1 = vz;
        const _Float16* kb = Kh + (size_t)(m0 + w * 16 + l16) * L_DIM + half * 8;
#pragma unroll
        for (int c = 0; c < 4; ++c) {
            H16 bf;
            bf.u[0] = *(const uint4*)(kb + c * 32);
            bf.u[1] = *(const uint4*)(kb + c * 32 + 16);
            d0 = WMMA_F16(qa[0][c], bf.h, d0);
            d1 = WMMA_F16(qa[1][c], bf.h, d1);
        }
        int col = w * 16 + l16;
#pragma unroll
        for (int r = 0; r < 8; ++r) {
            float s0 = fmaxf(d0[r], 0.f);
            float s1 = fmaxf(d1[r], 0.f);
            sq[0][r] += s0 * s0;
            sq[1][r] += s1 * s1;
            Sbuf[(r + 8 * half) * SSTR + col]      = (_Float16)s0;
            Sbuf[(16 + r + 8 * half) * SSTR + col] = (_Float16)s1;
        }
        __syncthreads();   // single barrier per tile (double-buffered S)

        // ---- phase 2: acc += S_f16 @ secondaryT
#pragma unroll
        for (int kk = 0; kk < 4; ++kk) {
            H16 a0, a1;
            const _Float16* s0p = &Sbuf[l16 * SSTR + kk * 32 + half * 8];
            const _Float16* s1p = &Sbuf[(16 + l16) * SSTR + kk * 32 + half * 8];
            a0.u[0] = *(const uint4*)(s0p); a0.u[1] = *(const uint4*)(s0p + 16);
            a1.u[0] = *(const uint4*)(s1p); a1.u[1] = *(const uint4*)(s1p + 16);
#pragma unroll
            for (int tc = 0; tc < 2; ++tc) {
                H16 bf;
                const _Float16* bp = ST + (size_t)(e0w + tc * 16 + l16) * M
                                        + m0 + kk * 32 + half * 8;
                bf.u[0] = *(const uint4*)(bp);
                bf.u[1] = *(const uint4*)(bp + 16);
                acc[0 + tc] = WMMA_F16(a0.h, bf.h, acc[0 + tc]);
                acc[2 + tc] = WMMA_F16(a1.h, bf.h, acc[2 + tc]);
            }
        }
    }

    // ---- one-time sumsq reduction (registers -> LDS)
#pragma unroll
    for (int ti = 0; ti < 2; ++ti)
#pragma unroll
        for (int r = 0; r < 8; ++r)
            atomicAdd(&sumsq[ti * 16 + r + 8 * half], sq[ti][r]);
    __syncthreads();

    // ---- normalize rows and store f32 output
#pragma unroll
    for (int ti = 0; ti < 2; ++ti) {
#pragma unroll
        for (int tc = 0; tc < 2; ++tc) {
            v8f c = acc[ti * 2 + tc];
#pragma unroll
            for (int r = 0; r < 8; ++r) {
                int row = ti * 16 + r + 8 * half;
                float nrm   = sqrtf(sumsq[row]);
                float scale = 1.f / fmaxf(nrm, EPSV);
                int e = e0w + tc * 16 + l16;
                Out[(size_t)(n0 + row) * E_DIM + e] = c[r] * scale;
            }
        }
    }
}

extern "C" void kernel_launch(void* const* d_in, const int* in_sizes, int n_in,
                              void* d_out, int out_size, void* d_ws, size_t ws_size,
                              hipStream_t stream)
{
    const float* primary   = (const float*)d_in[0];
    const float* secondary = (const float*)d_in[1];
    const float* W         = (const float*)d_in[2];
    const float* b         = (const float*)d_in[3];
    float* Out = (float*)d_out;

    int N = in_sizes[0] / E_DIM;   // 8192
    int M = in_sizes[1] / E_DIM;   // 8192

    char* ws = (char*)d_ws;
    _Float16* Q  = (_Float16*)(ws);                                          // N*L
    _Float16* K  = (_Float16*)(ws + (size_t)N * L_DIM * 2);                  // M*L
    _Float16* ST = (_Float16*)(ws + (size_t)(N + M) * L_DIM * 2);            // E*M
    _Float16* Wh = (_Float16*)(ws + (size_t)(N + M) * L_DIM * 2
                                  + (size_t)E_DIM * M * 2);                  // L*E

    cvt_w_kernel<<<(L_DIM * E_DIM + 255) / 256, 256, 0, stream>>>(W, Wh, L_DIM * E_DIM);
    transpose_cvt_kernel<<<dim3(M / 64, E_DIM / 64), 256, 0, stream>>>(secondary, ST, M);
    proj_kernel<<<(N + M) / 32, 256, 0, stream>>>(primary, secondary, Wh, b, Q, K, N, M);
    fused_attn_kernel<<<dim3(N / 32, E_DIM / 256), 256, 0, stream>>>(Q, K, ST, Out, N, M);
}
